// EVFN_naive_45664092291669
// MI455X (gfx1250) — matrix operand order repaired and checked by reading, compiled, not run on hardware
//
#include <hip/hip_runtime.h>
#include <hip/hip_bf16.h>

#define N_LAYERS 4

typedef __attribute__((ext_vector_type(16))) __bf16 v16bf;
typedef __attribute__((ext_vector_type(8)))  float  v8f;
typedef __attribute__((ext_vector_type(4)))  int    v4i;

__device__ __forceinline__ float silu_f(float x) { return x / (1.0f + __expf(-x)); }

// A-tile loader (16x32 bf16, M x K). Lane = row (mod 16); lanes 0-15 hold K {k0..k0+7, k0+16..k0+23},
// lanes 16-31 hold K {k0+8..k0+15, k0+24..k0+31}  (CDNA5 ISA 7.12.2).
__device__ __forceinline__ v16bf lds_tileA(const __bf16* s, int stride, int lane, int kbase) {
  int row = lane & 15;
  int k0  = kbase + ((lane & 16) ? 8 : 0);
  const __bf16* p = s + row * stride + k0;
  union { v4i q[2]; v16bf v; } u;
  u.q[0] = *(const v4i*)p;
  u.q[1] = *(const v4i*)(p + 16);
  return u.v;
}

// B-tile loader (32x16 bf16, K x N) from N-major (transposed) weights: lane = col (mod 16);
// lanes 0-15 hold K {k0..k0+7, k0+16..k0+23}, lanes 16-31 hold K {k0+8..k0+15, k0+24..k0+31}.
__device__ __forceinline__ v16bf lds_tileB(const __bf16* w, int stride, int lane, int kbase, int nbase) {
  int n  = nbase + (lane & 15);
  int k0 = kbase + ((lane & 16) ? 8 : 0);
  const __bf16* p = w + n * stride + k0;
  union { v4i q[2]; v16bf v; } u;
  u.q[0] = *(const v4i*)p;
  u.q[1] = *(const v4i*)(p + 16);
  return u.v;
}

// D-tile (f32 16x16): lane holds col n = lane&15, rows m = (lane>=16 ? 8:0)+j.
__device__ __forceinline__ void store_tile_bf(__bf16* s, int stride, int lane, int nbase, v8f d) {
  int n  = nbase + (lane & 15);
  int mb = (lane & 16) ? 8 : 0;
#pragma unroll
  for (int j = 0; j < 8; ++j) s[(mb + j) * stride + n] = (__bf16)d[j];
}

__device__ __forceinline__ void init_bias4(v8f* acc, const float* bias, int lane) {
  int nn = lane & 15;
#pragma unroll
  for (int t = 0; t < 4; ++t) {
    float b = bias[t * 16 + nn];
#pragma unroll
    for (int j = 0; j < 8; ++j) acc[t][j] = b;
  }
}

#define WMMA_BF16(a, b, c) \
  __builtin_amdgcn_wmma_f32_16x16x32_bf16(false, (a), false, (b), (short)0, (c), false, false)

// ---------------------------------------------------------------- small kernels

__global__ __launch_bounds__(256) void prep_weights(
    const float* __restrict__ e_w1, const float* __restrict__ e_w2,
    const float* __restrict__ c_w1, const float* __restrict__ n_w1,
    const float* __restrict__ n_w2, const float* __restrict__ v_w1,
    __bf16* __restrict__ EW1T, __bf16* __restrict__ EW2T, __bf16* __restrict__ CW1T,
    __bf16* __restrict__ NW1T, __bf16* __restrict__ NW2T, __bf16* __restrict__ VW1T) {
  int i = blockIdx.x * 256 + threadIdx.x;
  const int S_EW1 = N_LAYERS * 64 * 192;
  const int S_SQ  = N_LAYERS * 64 * 64;
  const int S_NW1 = N_LAYERS * 64 * 128;
  if (i < S_EW1) {
    // padded/permuted K: cols 0..127 = e_w1 rows 0..127 (h_row|h_col),
    // cols 128..159 = rows 129..160 (ef), col 160 = row 128 (rad), cols 161..191 = 0
    int l = i / (64 * 192), rem = i % (64 * 192);
    int j = rem / 192, k = rem % 192;
    int src = (k < 128) ? k : (k < 160 ? (k + 1) : (k == 160 ? 128 : -1));
    float v = (src >= 0) ? e_w1[(size_t)l * 161 * 64 + (size_t)src * 64 + j] : 0.0f;
    EW1T[i] = (__bf16)v;
    return;
  }
  i -= S_EW1;
  if (i < S_SQ) { int l=i/4096, rem=i%4096, j=rem/64, k=rem%64;
    EW2T[i] = (__bf16)e_w2[(size_t)l*4096 + (size_t)k*64 + j]; return; }
  i -= S_SQ;
  if (i < S_SQ) { int l=i/4096, rem=i%4096, j=rem/64, k=rem%64;
    CW1T[i] = (__bf16)c_w1[(size_t)l*4096 + (size_t)k*64 + j]; return; }
  i -= S_SQ;
  if (i < S_NW1) { int l=i/8192, rem=i%8192, j=rem/128, k=rem%128;
    NW1T[i] = (__bf16)n_w1[(size_t)l*8192 + (size_t)k*64 + j]; return; }
  i -= S_NW1;
  if (i < S_SQ) { int l=i/4096, rem=i%4096, j=rem/64, k=rem%64;
    NW2T[i] = (__bf16)n_w2[(size_t)l*4096 + (size_t)k*64 + j]; return; }
  i -= S_SQ;
  if (i < S_SQ) { int l=i/4096, rem=i%4096, j=rem/64, k=rem%64;
    VW1T[i] = (__bf16)v_w1[(size_t)l*4096 + (size_t)k*64 + j]; }
}

__global__ __launch_bounds__(256) void embed_kernel(
    const float* __restrict__ h_in, const float* __restrict__ w, const float* __restrict__ b,
    float* __restrict__ hg, __bf16* __restrict__ hbf, int N) {
  int i = blockIdx.x * 256 + threadIdx.x;
  if (i >= N * 64) return;
  int n = i >> 6, j = i & 63;
  float acc = b[j];
#pragma unroll
  for (int k = 0; k < 8; ++k) acc = fmaf(h_in[(size_t)n * 8 + k], w[k * 64 + j], acc);
  hg[i] = acc;
  hbf[i] = (__bf16)acc;
}

__global__ __launch_bounds__(256) void centroid_kernel(
    const float* __restrict__ x, float* __restrict__ xc, float* __restrict__ cent, int B) {
  int s = blockIdx.x * 256 + threadIdx.x;
  if (s >= B) return;
  float c0 = 0.f, c1 = 0.f, c2 = 0.f;
#pragma unroll
  for (int p = 0; p < 5; ++p) {
    size_t i = ((size_t)s * 5 + p) * 3;
    c0 += x[i]; c1 += x[i + 1]; c2 += x[i + 2];
  }
  c0 *= 0.2f; c1 *= 0.2f; c2 *= 0.2f;
  cent[(size_t)s * 3 + 0] = c0; cent[(size_t)s * 3 + 1] = c1; cent[(size_t)s * 3 + 2] = c2;
#pragma unroll
  for (int p = 0; p < 5; ++p) {
    size_t i = ((size_t)s * 5 + p) * 3;
    xc[i] = x[i] - c0; xc[i + 1] = x[i + 1] - c1; xc[i + 2] = x[i + 2] - c2;
  }
}

__global__ __launch_bounds__(256) void scalarize_fuse(
    const float* __restrict__ xc, const float* __restrict__ ea, const int* __restrict__ edges,
    const float* __restrict__ fw1, const float* __restrict__ fb1,
    const float* __restrict__ fw2, const float* __restrict__ fb2,
    __bf16* __restrict__ efbf, int E) {
  __shared__ float W1[10 * 32], B1[32], W2[32 * 32], B2[32];
  int tid = threadIdx.x;
  for (int i = tid; i < 320; i += 256) W1[i] = fw1[i];
  for (int i = tid; i < 1024; i += 256) W2[i] = fw2[i];
  if (tid < 32) { B1[tid] = fb1[tid]; B2[tid] = fb2[tid]; }
  __syncthreads();
  int e = blockIdx.x * 256 + tid;
  if (e >= E) return;
  int row = edges[e], col = edges[E + e];
  float a0 = xc[(size_t)row*3+0], a1 = xc[(size_t)row*3+1], a2 = xc[(size_t)row*3+2];
  float b0 = xc[(size_t)col*3+0], b1 = xc[(size_t)col*3+1], b2 = xc[(size_t)col*3+2];
  float dx = a0-b0, dy = a1-b1, dz = a2-b2;
  float radial = dx*dx + dy*dy + dz*dz;
  float cx = a1*b2 - a2*b1, cy = a2*b0 - a0*b2, cz = a0*b1 - a1*b0;
  float rn = 1.0f / (sqrtf(radial) + 1.0f);
  dx *= rn; dy *= rn; dz *= rn;
  float cn = 1.0f / (sqrtf(cx*cx + cy*cy + cz*cz) + 1.0f);
  cx *= cn; cy *= cn; cz *= cn;
  float vx = dy*cz - dz*cy, vy = dz*cx - dx*cz, vz = dx*cy - dy*cx;
  float ci0 = dx*a0 + dy*a1 + dz*a2;
  float ci1 = cx*a0 + cy*a1 + cz*a2;
  float ci2 = vx*a0 + vy*a1 + vz*a2;
  float cj0 = dx*b0 + dy*b1 + dz*b2;
  float cj1 = cx*b0 + cy*b1 + cz*b2;
  float cj2 = vx*b0 + vy*b1 + vz*b2;
  float ni = sqrtf(ci0*ci0 + ci1*ci1 + ci2*ci2);
  float nj = sqrtf(cj0*cj0 + cj1*cj1 + cj2*cj2);
  float cosv = (ci0*cj0 + ci1*cj1 + ci2*cj2) / ((ni + 1e-5f) * (nj + 1e-5f));
  float sinv = sqrtf(fmaxf(1.0f - cosv * cosv, 1e-12f));
  float feat[10] = { ea[(size_t)e*2+0], ea[(size_t)e*2+1], sinv, cosv,
                     ci0, ci1, ci2, cj0, cj1, cj2 };
  float t1[32];
#pragma unroll
  for (int j = 0; j < 32; ++j) {
    float acc = B1[j];
#pragma unroll
    for (int k = 0; k < 10; ++k) acc = fmaf(feat[k], W1[k * 32 + j], acc);
    t1[j] = silu_f(acc);
  }
#pragma unroll
  for (int j = 0; j < 32; ++j) {
    float acc = B2[j];
#pragma unroll
    for (int k = 0; k < 32; ++k) acc = fmaf(t1[k], W2[k * 32 + j], acc);
    efbf[(size_t)e * 32 + j] = (__bf16)silu_f(acc);
  }
}

__global__ __launch_bounds__(256) void fill_zero(float* __restrict__ p, int count) {
  int i = blockIdx.x * 256 + threadIdx.x;
  if (i < count) p[i] = 0.0f;
}

__global__ __launch_bounds__(256) void finalize_kernel(
    const float* __restrict__ xc, const float* __restrict__ cent, float* __restrict__ out, int N) {
  int i = blockIdx.x * 256 + threadIdx.x;
  if (i >= N * 3) return;
  int n = i / 3, c = i - n * 3;
  out[i] = xc[i] + cent[(size_t)(n / 5) * 3 + c];
}

// ---------------------------------------------------------------- edge kernel (WMMA)
// 64 threads = 2 waves; each wave owns 16 edges.
// edge MLP: [16,192]x[192,64] -> silu -> [16,64]x[64,64] -> silu = m
// coord MLP: silu(m x c_w1 + b) x c_w2, then scatter trans/agg/cnt + hagg.
__global__ __launch_bounds__(64) void edge_kernel(
    const float* __restrict__ xc, const __bf16* __restrict__ hbf,
    const __bf16* __restrict__ efbf, const int* __restrict__ edges, int E,
    const __bf16* __restrict__ W1Tg, const __bf16* __restrict__ W2Tg,
    const __bf16* __restrict__ CW1g,
    const float* __restrict__ b1g, const float* __restrict__ b2g,
    const float* __restrict__ cb1g, const float* __restrict__ cw2g,
    float* __restrict__ hagg, float* __restrict__ aggx, float* __restrict__ cnt) {
  __shared__ __attribute__((aligned(16))) __bf16 W1s[64 * 192];
  __shared__ __attribute__((aligned(16))) __bf16 W2s[64 * 64];
  __shared__ __attribute__((aligned(16))) __bf16 CW1s[64 * 64];
  __shared__ float B1s[64], B2s[64], CB1s[64], CW2s[192];
  __shared__ __attribute__((aligned(16))) __bf16 S0[2][16 * 192];
  __shared__ __attribute__((aligned(16))) __bf16 S1[2][16 * 64];
  __shared__ int RIDX[2][16];
  __shared__ float XA[2][16][3], XB[2][16][3];

  int tid = threadIdx.x;
  for (int i = tid; i < (64 * 192) / 8; i += 64) ((v4i*)W1s)[i] = ((const v4i*)W1Tg)[i];
  for (int i = tid; i < (64 * 64) / 8; i += 64) {
    ((v4i*)W2s)[i]  = ((const v4i*)W2Tg)[i];
    ((v4i*)CW1s)[i] = ((const v4i*)CW1g)[i];
  }
  if (tid < 64) { B1s[tid] = b1g[tid]; B2s[tid] = b2g[tid]; CB1s[tid] = cb1g[tid]; }
  for (int i = tid; i < 192; i += 64) CW2s[i] = cw2g[i];
  __syncthreads();

  int lane = tid & 31, wave = tid >> 5;
  int r = lane & 15, hsel = lane >> 4;
  int e0 = blockIdx.x * 32 + wave * 16;
  int eg = e0 + r;
  int e  = (eg < E) ? eg : (E - 1);
  int row = edges[e], col = edges[E + e];
  __bf16* s0 = S0[wave];
  __bf16* s1 = S1[wave];

  // assemble A = [h_row(0:64) | h_col(64:128) | ef(128:160) | rad(160) | 0(161:192)]
  if (hsel == 0) {
    RIDX[wave][r] = row;
    const v4i* hp = (const v4i*)(hbf + (size_t)row * 64);
    v4i* d0 = (v4i*)(s0 + r * 192);
#pragma unroll
    for (int i = 0; i < 8; ++i) d0[i] = hp[i];
    const v4i* cp = (const v4i*)(hbf + (size_t)col * 64);
    v4i* d1 = (v4i*)(s0 + r * 192 + 64);
#pragma unroll
    for (int i = 0; i < 4; ++i) d1[i] = cp[i];
  } else {
    const v4i* cp = (const v4i*)(hbf + (size_t)col * 64 + 32);
    v4i* d1 = (v4i*)(s0 + r * 192 + 96);
#pragma unroll
    for (int i = 0; i < 4; ++i) d1[i] = cp[i];
    const v4i* ep = (const v4i*)(efbf + (size_t)e * 32);
    v4i* d2 = (v4i*)(s0 + r * 192 + 128);
#pragma unroll
    for (int i = 0; i < 4; ++i) d2[i] = ep[i];
    float a0 = xc[(size_t)row*3+0], a1 = xc[(size_t)row*3+1], a2 = xc[(size_t)row*3+2];
    float b0 = xc[(size_t)col*3+0], b1 = xc[(size_t)col*3+1], b2 = xc[(size_t)col*3+2];
    XA[wave][r][0] = a0; XA[wave][r][1] = a1; XA[wave][r][2] = a2;
    XB[wave][r][0] = b0; XB[wave][r][1] = b1; XB[wave][r][2] = b2;
    float dx = a0 - b0, dy = a1 - b1, dz = a2 - b2;
    s0[r * 192 + 160] = (__bf16)(dx * dx + dy * dy + dz * dz);
#pragma unroll
    for (int k = 161; k < 192; ++k) s0[r * 192 + k] = (__bf16)0.0f;
  }

  // stage 1: silu(m_in @ e_w1 + b1) -> S1
  v8f a1[4];
  init_bias4(a1, B1s, lane);
#pragma unroll
  for (int kc = 0; kc < 6; ++kc) {
    v16bf A = lds_tileA(s0, 192, lane, kc * 32);
#pragma unroll
    for (int t = 0; t < 4; ++t)
      a1[t] = WMMA_BF16(A, lds_tileB(W1s, 192, lane, kc * 32, t * 16), a1[t]);
  }
#pragma unroll
  for (int t = 0; t < 4; ++t) {
#pragma unroll
    for (int j = 0; j < 8; ++j) a1[t][j] = silu_f(a1[t][j]);
    store_tile_bf(s1, 64, lane, t * 16, a1[t]);
  }

  // stage 2: m = silu(S1 @ e_w2 + b2) -> S0 (first 16x64), scatter into hagg
  v8f a2[4];
  init_bias4(a2, B2s, lane);
#pragma unroll
  for (int kc = 0; kc < 2; ++kc) {
    v16bf A = lds_tileA(s1, 64, lane, kc * 32);
#pragma unroll
    for (int t = 0; t < 4; ++t)
      a2[t] = WMMA_BF16(A, lds_tileB(W2s, 64, lane, kc * 32, t * 16), a2[t]);
  }
  {
    int nn = lane & 15, mb = (lane & 16) ? 8 : 0;
#pragma unroll
    for (int t = 0; t < 4; ++t)
#pragma unroll
      for (int j = 0; j < 8; ++j) {
        float mv = silu_f(a2[t][j]);
        int mrow = mb + j;
        s0[mrow * 64 + t * 16 + nn] = (__bf16)mv;
        if (e0 + mrow < E)
          atomicAdd(hagg + (size_t)RIDX[wave][mrow] * 64 + t * 16 + nn, mv);
      }
  }

  // stage 3: T = silu(m @ c_w1 + cb1) -> S1
  v8f a3[4];
  init_bias4(a3, CB1s, lane);
#pragma unroll
  for (int kc = 0; kc < 2; ++kc) {
    v16bf A = lds_tileA(s0, 64, lane, kc * 32);
#pragma unroll
    for (int t = 0; t < 4; ++t)
      a3[t] = WMMA_BF16(A, lds_tileB(CW1s, 64, lane, kc * 32, t * 16), a3[t]);
  }
#pragma unroll
  for (int t = 0; t < 4; ++t) {
#pragma unroll
    for (int j = 0; j < 8; ++j) a3[t][j] = silu_f(a3[t][j]);
    store_tile_bf(s1, 64, lane, t * 16, a3[t]);
  }

  // stage 4: coff = T @ c_w2 (E x 3), build trans, scatter agg/cnt
  if (lane < 16 && (e0 + lane) < E) {
    float c0 = 0.f, c1 = 0.f, c2 = 0.f;
#pragma unroll
    for (int k = 0; k < 64; ++k) {
      float tv = (float)s1[lane * 64 + k];
      c0 = fmaf(tv, CW2s[k * 3 + 0], c0);
      c1 = fmaf(tv, CW2s[k * 3 + 1], c1);
      c2 = fmaf(tv, CW2s[k * 3 + 2], c2);
    }
    float a0 = XA[wave][lane][0], a1v = XA[wave][lane][1], a2v = XA[wave][lane][2];
    float b0 = XB[wave][lane][0], b1v = XB[wave][lane][1], b2v = XB[wave][lane][2];
    float dx = a0 - b0, dy = a1v - b1v, dz = a2v - b2v;
    float cx = a1v * b2v - a2v * b1v, cy = a2v * b0 - a0 * b2v, cz = a0 * b1v - a1v * b0;
    float vx = dy * cz - dz * cy, vy = dz * cx - dx * cz, vz = dx * cy - dy * cx;
    float t0 = dx * c0 + cx * c1 + vx * c2;
    float t1 = dy * c0 + cy * c1 + vy * c2;
    float t2 = dz * c0 + cz * c1 + vz * c2;
    int rw = RIDX[wave][lane];
    atomicAdd(aggx + (size_t)rw * 3 + 0, t0);
    atomicAdd(aggx + (size_t)rw * 3 + 1, t1);
    atomicAdd(aggx + (size_t)rw * 3 + 2, t2);
    atomicAdd(cnt + rw, 1.0f);
  }
}

// ---------------------------------------------------------------- node kernel (WMMA)
// node MLP: silu([h|hagg] @ n_w1 + b) @ n_w2 + b2 (residual into h);
// vel MLP: (silu(h @ v_w1 + b) @ v_w2 + b2) * vel; coord update.
__global__ __launch_bounds__(64) void node_kernel(
    float* __restrict__ hg, __bf16* __restrict__ hbf,
    const float* __restrict__ hagg, const float* __restrict__ aggx,
    const float* __restrict__ cnt, float* __restrict__ xc,
    const float* __restrict__ vel, int N,
    const __bf16* __restrict__ NW1g, const __bf16* __restrict__ NW2g,
    const __bf16* __restrict__ VW1g,
    const float* __restrict__ nb1g, const float* __restrict__ nb2g,
    const float* __restrict__ vb1g, const float* __restrict__ vw2g,
    const float* __restrict__ vb2g) {
  __shared__ __attribute__((aligned(16))) __bf16 NW1s[64 * 128];
  __shared__ __attribute__((aligned(16))) __bf16 NW2s[64 * 64];
  __shared__ __attribute__((aligned(16))) __bf16 VW1s[64 * 64];
  __shared__ float NB1s[64], NB2s[64], VB1s[64], VW2s[64];
  __shared__ __attribute__((aligned(16))) __bf16 S0[2][16 * 128];
  __shared__ __attribute__((aligned(16))) __bf16 S1[2][16 * 64];

  int tid = threadIdx.x;
  for (int i = tid; i < (64 * 128) / 8; i += 64) ((v4i*)NW1s)[i] = ((const v4i*)NW1g)[i];
  for (int i = tid; i < (64 * 64) / 8; i += 64) {
    ((v4i*)NW2s)[i] = ((const v4i*)NW2g)[i];
    ((v4i*)VW1s)[i] = ((const v4i*)VW1g)[i];
  }
  if (tid < 64) { NB1s[tid] = nb1g[tid]; NB2s[tid] = nb2g[tid]; VB1s[tid] = vb1g[tid]; VW2s[tid] = vw2g[tid]; }
  __syncthreads();

  int lane = tid & 31, wave = tid >> 5;
  int r = lane & 15, hsel = lane >> 4;
  int n0 = blockIdx.x * 32 + wave * 16;
  int ng = n0 + r;
  int n  = (ng < N) ? ng : (N - 1);
  __bf16* s0 = S0[wave];
  __bf16* s1 = S1[wave];

  if (hsel == 0) {
    const v4i* hp = (const v4i*)(hbf + (size_t)n * 64);
    v4i* d0 = (v4i*)(s0 + r * 128);
#pragma unroll
    for (int i = 0; i < 8; ++i) d0[i] = hp[i];
  } else {
    const float* ap = hagg + (size_t)n * 64;
#pragma unroll
    for (int k = 0; k < 64; ++k) s0[r * 128 + 64 + k] = (__bf16)ap[k];
  }

  // stage 1: tmp = silu([h|hagg] @ n_w1 + nb1) -> S1
  v8f a1[4];
  init_bias4(a1, NB1s, lane);
#pragma unroll
  for (int kc = 0; kc < 4; ++kc) {
    v16bf A = lds_tileA(s0, 128, lane, kc * 32);
#pragma unroll
    for (int t = 0; t < 4; ++t)
      a1[t] = WMMA_BF16(A, lds_tileB(NW1s, 128, lane, kc * 32, t * 16), a1[t]);
  }
#pragma unroll
  for (int t = 0; t < 4; ++t) {
#pragma unroll
    for (int j = 0; j < 8; ++j) a1[t][j] = silu_f(a1[t][j]);
    store_tile_bf(s1, 64, lane, t * 16, a1[t]);
  }

  // stage 2: hout = tmp @ n_w2 + nb2 (kept in registers)
  v8f a2[4];
  init_bias4(a2, NB2s, lane);
#pragma unroll
  for (int kc = 0; kc < 2; ++kc) {
    v16bf A = lds_tileA(s1, 64, lane, kc * 32);
#pragma unroll
    for (int t = 0; t < 4; ++t)
      a2[t] = WMMA_BF16(A, lds_tileB(NW2s, 64, lane, kc * 32, t * 16), a2[t]);
  }

  // stage 3: vt = silu(h @ v_w1 + vb1) -> S1 (h = first 64 cols of S0)
  v8f a3[4];
  init_bias4(a3, VB1s, lane);
#pragma unroll
  for (int kc = 0; kc < 2; ++kc) {
    v16bf A = lds_tileA(s0, 128, lane, kc * 32);
#pragma unroll
    for (int t = 0; t < 4; ++t)
      a3[t] = WMMA_BF16(A, lds_tileB(VW1s, 64, lane, kc * 32, t * 16), a3[t]);
  }
#pragma unroll
  for (int t = 0; t < 4; ++t) {
#pragma unroll
    for (int j = 0; j < 8; ++j) a3[t][j] = silu_f(a3[t][j]);
    store_tile_bf(s1, 64, lane, t * 16, a3[t]);
  }

  // h residual update
  {
    int nn = lane & 15, mb = (lane & 16) ? 8 : 0;
#pragma unroll
    for (int t = 0; t < 4; ++t)
#pragma unroll
      for (int j = 0; j < 8; ++j) {
        int nd = n0 + mb + j;
        if (nd < N) {
          size_t idx = (size_t)nd * 64 + t * 16 + nn;
          float nh = hg[idx] + a2[t][j];
          hg[idx] = nh;
          hbf[idx] = (__bf16)nh;
        }
      }
  }

  // coord update: xc += agg/max(cnt,1) + (vt @ v_w2 + vb2) * vel
  if (lane < 16 && (n0 + lane) < N) {
    int nd = n0 + lane;
    float vs = vb2g[0];
#pragma unroll
    for (int k = 0; k < 64; ++k) vs = fmaf((float)s1[lane * 64 + k], VW2s[k], vs);
    float inv = 1.0f / fmaxf(cnt[nd], 1.0f);
#pragma unroll
    for (int c = 0; c < 3; ++c) {
      size_t i3 = (size_t)nd * 3 + c;
      xc[i3] = xc[i3] + aggx[i3] * inv + vs * vel[i3];
    }
  }
}

// ---------------------------------------------------------------- host launcher

extern "C" void kernel_launch(void* const* d_in, const int* in_sizes, int n_in,
                              void* d_out, int out_size, void* d_ws, size_t ws_size,
                              hipStream_t stream) {
  const float* h_in  = (const float*)d_in[0];
  const float* x_in  = (const float*)d_in[1];
  const float* vel   = (const float*)d_in[2];
  const float* eattr = (const float*)d_in[3];
  const int*   edges = (const int*)d_in[4];
  const float* emb_w = (const float*)d_in[5];
  const float* emb_b = (const float*)d_in[6];
  const float* fw1 = (const float*)d_in[7];
  const float* fb1 = (const float*)d_in[8];
  const float* fw2 = (const float*)d_in[9];
  const float* fb2 = (const float*)d_in[10];
  const float* e_w1 = (const float*)d_in[11];
  const float* e_b1 = (const float*)d_in[12];
  const float* e_w2 = (const float*)d_in[13];
  const float* e_b2 = (const float*)d_in[14];
  const float* n_w1 = (const float*)d_in[15];
  const float* n_b1 = (const float*)d_in[16];
  const float* n_w2 = (const float*)d_in[17];
  const float* n_b2 = (const float*)d_in[18];
  const float* c_w1 = (const float*)d_in[19];
  const float* c_b1 = (const float*)d_in[20];
  const float* c_w2 = (const float*)d_in[21];
  const float* v_w1 = (const float*)d_in[22];
  const float* v_b1 = (const float*)d_in[23];
  const float* v_w2 = (const float*)d_in[24];
  const float* v_b2 = (const float*)d_in[25];

  int N = in_sizes[0] / 8;
  int E = in_sizes[3] / 2;
  int B = N / 5;
  (void)n_in; (void)ws_size;

  // workspace carve (256B aligned)
  char* w = (char*)d_ws;
  size_t off = 0;
  auto carve = [&](size_t bytes) { size_t o = off; off = (off + bytes + 255) & ~(size_t)255; return o; };
  float*  hg    = (float*) (w + carve((size_t)N * 64 * 4));
  __bf16* hbf   = (__bf16*)(w + carve((size_t)N * 64 * 2));
  float*  xcg   = (float*) (w + carve((size_t)N * 3 * 4));
  float*  centg = (float*) (w + carve((size_t)B * 3 * 4));
  __bf16* efbf  = (__bf16*)(w + carve((size_t)E * 32 * 2));
  float*  hagg  = (float*) (w + carve((size_t)N * 68 * 4));  // hagg | aggx | cnt contiguous
  float*  aggx  = hagg + (size_t)N * 64;
  float*  cntg  = aggx + (size_t)N * 3;
  __bf16* EW1T  = (__bf16*)(w + carve((size_t)N_LAYERS * 64 * 192 * 2));
  __bf16* EW2T  = (__bf16*)(w + carve((size_t)N_LAYERS * 64 * 64 * 2));
  __bf16* CW1T  = (__bf16*)(w + carve((size_t)N_LAYERS * 64 * 64 * 2));
  __bf16* NW1T  = (__bf16*)(w + carve((size_t)N_LAYERS * 64 * 128 * 2));
  __bf16* NW2T  = (__bf16*)(w + carve((size_t)N_LAYERS * 64 * 64 * 2));
  __bf16* VW1T  = (__bf16*)(w + carve((size_t)N_LAYERS * 64 * 64 * 2));

  const int prepTotal = N_LAYERS * (64 * 192 + 4 * 64 * 64 + 64 * 128);
  prep_weights<<<(prepTotal + 255) / 256, 256, 0, stream>>>(
      e_w1, e_w2, c_w1, n_w1, n_w2, v_w1, EW1T, EW2T, CW1T, NW1T, NW2T, VW1T);

  embed_kernel<<<(N * 64 + 255) / 256, 256, 0, stream>>>(h_in, emb_w, emb_b, hg, hbf, N);
  centroid_kernel<<<(B + 255) / 256, 256, 0, stream>>>(x_in, xcg, centg, B);
  scalarize_fuse<<<(E + 255) / 256, 256, 0, stream>>>(xcg, eattr, edges, fw1, fb1, fw2, fb2, efbf, E);

  for (int l = 0; l < N_LAYERS; ++l) {
    fill_zero<<<(N * 68 + 255) / 256, 256, 0, stream>>>(hagg, N * 68);
    edge_kernel<<<(E + 31) / 32, 64, 0, stream>>>(
        xcg, hbf, efbf, edges, E,
        EW1T + (size_t)l * 64 * 192, EW2T + (size_t)l * 4096, CW1T + (size_t)l * 4096,
        e_b1 + (size_t)l * 64, e_b2 + (size_t)l * 64, c_b1 + (size_t)l * 64, c_w2 + (size_t)l * 192,
        hagg, aggx, cntg);
    node_kernel<<<(N + 31) / 32, 64, 0, stream>>>(
        hg, hbf, hagg, aggx, cntg, xcg, vel, N,
        NW1T + (size_t)l * 8192, NW2T + (size_t)l * 4096, VW1T + (size_t)l * 4096,
        n_b1 + (size_t)l * 64, n_b2 + (size_t)l * 64,
        v_b1 + (size_t)l * 64, v_w2 + (size_t)l * 64, v_b2 + l);
  }

  finalize_kernel<<<(N * 3 + 255) / 256, 256, 0, stream>>>(xcg, centg, (float*)d_out, N);
}
